// GINConvClassifier_26302379720855
// MI455X (gfx1250) — matrix operand order repaired and checked by reading, not compiled, not run on hardware
//
#include <hip/hip_runtime.h>
#include <hip/hip_bf16.h>

// ---------------------------------------------------------------------------
// GIN classifier for MI455X (gfx1250, wave32, WMMA).
// Bandwidth-bound problem (edge scatter ~1.2 GB random traffic dominates);
// GEMMs (64x64) run on v_wmma_f32_16x16x32_f16 with fp32 accumulation.
// ---------------------------------------------------------------------------

#define N_NODES    100000
#define N_EDGES    1600000
#define CH         64
#define NUM_GRAPHS 2048
#define N_LAYERS   3
#define BN_EPS     1e-5f

typedef __attribute__((ext_vector_type(16))) _Float16 v16h;
typedef __attribute__((ext_vector_type(8)))  float    v8f;

// ---- WMMA fragment helpers (wave32; A and B share contiguous k = 16*half + i) ----------

// A fragment (16x32 f16) from an f32 row-major matrix (row stride in floats).
// lane: m = lane&15 (row), half = lane>>4. Element i -> K = kb + half*16 + i.
__device__ __forceinline__ v16h a_frag_from_f32(const float* __restrict__ base,
                                                int row_stride, int m, int half,
                                                int kbase, float scale) {
  const float* p = base + (size_t)m * row_stride + kbase + half * 16;
  v16h a;
#pragma unroll
  for (int i = 0; i < 8; ++i) {
    a[i]     = (_Float16)(p[i] * scale);
    a[8 + i] = (_Float16)(p[8 + i] * scale);
  }
  return a;
}

// A fragment from f16 LDS tile (row stride in halfs; stride chosen 16B aligned).
__device__ __forceinline__ v16h a_frag_from_lds(const _Float16* __restrict__ lds,
                                                int row_stride, int m, int half,
                                                int kbase) {
  const _Float16* p = lds + m * row_stride + kbase + half * 16;
  v16h a;
#pragma unroll
  for (int i = 0; i < 8; ++i) {
    a[i]     = p[i];
    a[8 + i] = p[8 + i];
  }
  return a;
}

// B fragment (32x16 f16) from pre-transposed weights Wt[N][K] (K stride 64).
// lane: n = lane&15 (col), half = lane>>4. Element i -> K = kbase + half*16 + i.
__device__ __forceinline__ v16h b_frag(const _Float16* __restrict__ Wt,
                                       int n, int half, int kbase) {
  const _Float16* p = Wt + (size_t)n * CH + kbase + half * 16;
  v16h b;
#pragma unroll
  for (int i = 0; i < 16; ++i) b[i] = p[i];
  return b;
}

__device__ __forceinline__ v8f wmma_f16(v16h a, v16h b, v8f c) {
  return __builtin_amdgcn_wmma_f32_16x16x32_f16(false, a, false, b,
                                                (short)0, c, false, false);
}

// ---- prep: transpose weights to [N][K] f16, fold BN+bias1 into scale/shift -
__global__ __launch_bounds__(256) void gin_prep_kernel(
    const float* __restrict__ convW1, const float* __restrict__ convW2,
    const float* __restrict__ mlpW1,  const float* __restrict__ convb1,
    const float* __restrict__ gamma,  const float* __restrict__ beta,
    const float* __restrict__ mean,   const float* __restrict__ var,
    _Float16* __restrict__ W1t, _Float16* __restrict__ W2t,
    _Float16* __restrict__ M1t, float* __restrict__ bnA, float* __restrict__ bnB) {
  int t = blockIdx.x * 256 + threadIdx.x;
  if (t < N_LAYERS * CH * CH) {              // Wt[l][n][k] = W[l][k][n]
    int l = t >> 12, n = (t >> 6) & 63, k = t & 63;
    int s = (l << 12) + (k << 6) + n;
    W1t[t] = (_Float16)convW1[s];
    W2t[t] = (_Float16)convW2[s];
  }
  if (t < CH * CH) {
    int n = t >> 6, k = t & 63;
    M1t[t] = (_Float16)mlpW1[(k << 6) + n];
  }
  if (t < N_LAYERS * CH) {                   // BN fold: y = a*lin + b
    float a = gamma[t] * rsqrtf(var[t] + BN_EPS);
    bnA[t] = a;
    bnB[t] = (convb1[t] - mean[t]) * a + beta[t];
  }
}

__global__ __launch_bounds__(256) void gin_zero_kernel(float* __restrict__ p, int n) {
  int t = blockIdx.x * 256 + threadIdx.x;
  if (t < n) p[t] = 0.0f;
}

__global__ __launch_bounds__(256) void gin_copy_kernel(float4* __restrict__ dst,
                                                       const float4* __restrict__ src,
                                                       int n4) {
  int t = blockIdx.x * 256 + threadIdx.x;
  if (t < n4) dst[t] = src[t];
}

// ---- edge scatter: h[dst] += x[src], 16 threads/edge, float4 per thread ----
__global__ __launch_bounds__(256) void gin_scatter_kernel(
    float* __restrict__ h, const float* __restrict__ x,
    const int* __restrict__ ei) {
  int t  = blockIdx.x * 256 + threadIdx.x;
  int e  = t >> 4;
  int c4 = (t & 15) * 4;
  if (e >= N_EDGES) return;
  int s = ei[e];
  int d = ei[N_EDGES + e];
  float4 v = *(const float4*)(x + (size_t)s * CH + c4);
  float* p = h + (size_t)d * CH + c4;
  atomicAdd(p + 0, v.x);
  atomicAdd(p + 1, v.y);
  atomicAdd(p + 2, v.z);
  atomicAdd(p + 3, v.w);
}

// ---- fused node MLP: (GEMM1 -> BN -> ReLU -> GEMM2 -> +b2 -> ReLU), inplace -
// 256 thr = 8 waves; wave (m_tile = w>>2, n_tile = w&3); 32 nodes per block.
// Grid sized exactly (100000 % 32 == 0) so no wave diverges before barrier.
#define LDSS 80  // f16 row stride: 160 B, multiple of 16 B, bank-skewed
__global__ __launch_bounds__(256) void gin_mlp_kernel(
    float* __restrict__ h, const _Float16* __restrict__ W1t,
    const _Float16* __restrict__ W2t, const float* __restrict__ bnA,
    const float* __restrict__ bnB, const float* __restrict__ b2) {
  __shared__ _Float16 lds_h[2][16 * LDSS];

  int tid    = threadIdx.x;
  int wave   = tid >> 5, lane = tid & 31;
  int m_tile = wave >> 2, n_tile = wave & 3;
  int half   = lane >> 4;
  int mn     = lane & 15;                 // A row within tile == C/D column
  int node0  = blockIdx.x * 32 + m_tile * 16;
  int col    = n_tile * 16 + mn;          // output channel of this lane

  const float* arow = h + (size_t)node0 * CH;

  // GEMM1: h[16x64] @ W1[64x64] (this wave: 16x16 tile, K = 64 = 2 WMMAs)
  v8f acc = {};
#pragma unroll
  for (int kb = 0; kb < 2; ++kb) {
    v16h a = a_frag_from_f32(arow, CH, mn, half, kb * 32, 1.0f);
    v16h b = b_frag(W1t, col, half, kb * 32);
    acc = wmma_f16(a, b, acc);
  }

  // BN(eval)+ReLU, stash as f16 into LDS (A-operand staging for GEMM2)
  float sA = bnA[col], sB = bnB[col];
  _Float16* lt = lds_h[m_tile];
#pragma unroll
  for (int r = 0; r < 8; ++r) {
    float v = acc[r] * sA + sB;
    v = v > 0.0f ? v : 0.0f;
    lt[(r + half * 8) * LDSS + col] = (_Float16)v;
  }
  __syncthreads();

  // GEMM2: relu(h1)[16x64] @ W2[64x64]
  v8f acc2 = {};
#pragma unroll
  for (int kb = 0; kb < 2; ++kb) {
    v16h a = a_frag_from_lds(lds_h[m_tile], LDSS, mn, half, kb * 32);
    v16h b = b_frag(W2t, col, half, kb * 32);
    acc2 = wmma_f16(a, b, acc2);
  }

  float bias = b2[col];
  float* orow = h + (size_t)node0 * CH;   // in place: block-private rows
#pragma unroll
  for (int r = 0; r < 8; ++r) {
    float v = acc2[r] + bias;
    orow[(size_t)(r + half * 8) * CH + col] = v > 0.0f ? v : 0.0f;
  }
}

// ---- graph mean-pool (sums + counts via atomics) ---------------------------
__global__ __launch_bounds__(256) void gin_pool_kernel(
    const float* __restrict__ x, const int* __restrict__ batch,
    float* __restrict__ sums, float* __restrict__ counts) {
  int t    = blockIdx.x * 256 + threadIdx.x;
  int node = t >> 4;
  int c4   = (t & 15) * 4;
  if (node >= N_NODES) return;
  int g = batch[node];
  float4 v = *(const float4*)(x + (size_t)node * CH + c4);
  float* p = sums + (size_t)g * CH + c4;
  atomicAdd(p + 0, v.x);
  atomicAdd(p + 1, v.y);
  atomicAdd(p + 2, v.z);
  atomicAdd(p + 3, v.w);
  if (c4 == 0) atomicAdd(&counts[g], 1.0f);
}

// ---- classifier: mean -> WMMA 64x64 -> ReLU -> 64x2 dot -> softmax ---------
// 128 thr = 4 waves, 16 graphs per block (2048 % 16 == 0).
__global__ __launch_bounds__(128) void gin_classify_kernel(
    const float* __restrict__ sums, const float* __restrict__ counts,
    const _Float16* __restrict__ M1t, const float* __restrict__ b1,
    const float* __restrict__ W2, const float* __restrict__ b2,
    float* __restrict__ out) {
  __shared__ float hid[16][CH + 8];

  int tid  = threadIdx.x;
  int wave = tid >> 5, lane = tid & 31;
  int half = lane >> 4, mn = lane & 15;
  int g0   = blockIdx.x * 16;

  float cnt = counts[g0 + mn];
  float inv = 1.0f / (cnt < 1.0f ? 1.0f : cnt);   // mean folded into A cvt

  v8f acc = {};
#pragma unroll
  for (int kb = 0; kb < 2; ++kb) {
    v16h a = a_frag_from_f32(sums + (size_t)g0 * CH, CH, mn, half, kb * 32, inv);
    v16h b = b_frag(M1t, wave * 16 + mn, half, kb * 32);
    acc = wmma_f16(a, b, acc);
  }
  int col = wave * 16 + mn;
  float bias = b1[col];
#pragma unroll
  for (int r = 0; r < 8; ++r) {
    float v = acc[r] + bias;
    hid[r + half * 8][col] = v > 0.0f ? v : 0.0f;
  }
  __syncthreads();

  if (tid < 16) {
    int g = g0 + tid;
    float l0 = b2[0], l1 = b2[1];
#pragma unroll 8
    for (int k = 0; k < CH; ++k) {
      float hv = hid[tid][k];
      l0 += hv * W2[k * 2 + 0];
      l1 += hv * W2[k * 2 + 1];
    }
    float mx = fmaxf(l0, l1);
    float e0 = __expf(l0 - mx), e1 = __expf(l1 - mx);
    float s  = 1.0f / (e0 + e1);
    out[(size_t)g * 2 + 0] = e0 * s;
    out[(size_t)g * 2 + 1] = e1 * s;
  }
}

// ---------------------------------------------------------------------------
extern "C" void kernel_launch(void* const* d_in, const int* in_sizes, int n_in,
                              void* d_out, int out_size, void* d_ws, size_t ws_size,
                              hipStream_t stream) {
  (void)in_sizes; (void)n_in; (void)out_size; (void)ws_size;

  const float* x      = (const float*)d_in[0];
  const int*   ei     = (const int*)d_in[1];
  const int*   batch  = (const int*)d_in[2];
  const float* convW1 = (const float*)d_in[3];
  const float* convb1 = (const float*)d_in[4];
  const float* gamma  = (const float*)d_in[5];
  const float* beta   = (const float*)d_in[6];
  const float* mean   = (const float*)d_in[7];
  const float* var    = (const float*)d_in[8];
  const float* convW2 = (const float*)d_in[9];
  const float* convb2 = (const float*)d_in[10];
  const float* mlpW1  = (const float*)d_in[11];
  const float* mlpb1  = (const float*)d_in[12];
  const float* mlpW2  = (const float*)d_in[13];
  const float* mlpb2  = (const float*)d_in[14];

  // workspace carve-out (all buffers written before read each call)
  char* ws = (char*)d_ws;
  size_t off = 0;
  auto carve = [&](size_t bytes) {
    void* p = ws + off;
    off = (off + bytes + 255) & ~(size_t)255;
    return p;
  };
  float*    buf0   = (float*)carve((size_t)N_NODES * CH * 4);    // 25.6 MB
  float*    buf1   = (float*)carve((size_t)N_NODES * CH * 4);    // 25.6 MB
  float*    sums   = (float*)carve((size_t)NUM_GRAPHS * CH * 4); // 512 KB
  float*    counts = (float*)carve((size_t)NUM_GRAPHS * 4);
  _Float16* W1t    = (_Float16*)carve((size_t)N_LAYERS * CH * CH * 2);
  _Float16* W2t    = (_Float16*)carve((size_t)N_LAYERS * CH * CH * 2);
  _Float16* M1t    = (_Float16*)carve((size_t)CH * CH * 2);
  float*    bnA    = (float*)carve((size_t)N_LAYERS * CH * 4);
  float*    bnB    = (float*)carve((size_t)N_LAYERS * CH * 4);

  // weight prep + pooled accumulator init
  gin_prep_kernel<<<48, 256, 0, stream>>>(convW1, convW2, mlpW1, convb1, gamma,
                                          beta, mean, var, W1t, W2t, M1t, bnA, bnB);
  gin_zero_kernel<<<(NUM_GRAPHS * CH + 255) / 256, 256, 0, stream>>>(sums, NUM_GRAPHS * CH);
  gin_zero_kernel<<<(NUM_GRAPHS + 255) / 256, 256, 0, stream>>>(counts, NUM_GRAPHS);

  const int n4        = N_NODES * CH / 4;           // 1.6M float4
  const int copy_grid = n4 / 256;                   // 6250
  const int scat_grid = N_EDGES * 16 / 256;         // 100000
  const int mlp_grid  = N_NODES / 32;               // 3125 (exact)

  const float* xs[4] = {x, buf0, buf1, buf0};       // layer inputs
  float*       hs[3] = {buf0, buf1, buf0};          // layer h/output buffers

  for (int l = 0; l < N_LAYERS; ++l) {
    gin_copy_kernel<<<copy_grid, 256, 0, stream>>>((float4*)hs[l], (const float4*)xs[l], n4);
    gin_scatter_kernel<<<scat_grid, 256, 0, stream>>>(hs[l], xs[l], ei);
    gin_mlp_kernel<<<mlp_grid, 256, 0, stream>>>(hs[l],
                                                 W1t + (size_t)l * CH * CH,
                                                 W2t + (size_t)l * CH * CH,
                                                 bnA + l * CH, bnB + l * CH,
                                                 convb2 + l * CH);
  }

  gin_pool_kernel<<<N_NODES * 16 / 256, 256, 0, stream>>>(buf0, batch, sums, counts);
  gin_classify_kernel<<<NUM_GRAPHS / 16, 128, 0, stream>>>(sums, counts, M1t, mlpb1,
                                                           mlpW2, mlpb2, (float*)d_out);
}
